// GptOssAttention_19095424598731
// MI455X (gfx1250) — compile-verified, matrix operand order checked by reading, not run on hardware
//
#include <hip/hip_runtime.h>

// ---------------------------------------------------------------------------
// GPT-OSS sliding-window attention layer for gfx1250 (MI455X).
// bf16 WMMA (v_wmma_f32_16x16x32_bf16) for both projection GEMMs and both
// attention matmuls; f32 softmax/RoPE/bias epilogues.
// Tile staging uses CDNA5 async global->LDS DMA (global_load_async_to_lds_b128,
// ASYNCcnt) with double buffering in the GEMM.
// ---------------------------------------------------------------------------

typedef __attribute__((ext_vector_type(16))) __bf16 v16bf;
typedef __attribute__((ext_vector_type(8)))  float  v8f;

union Frag {
    v16bf v;
    uint4 u[2];
};

#define H_   64
#define HKV_ 8
#define D_   64
#define W_   128
#define S_   2048
#define HID_ 2880
#define NQKV 5120   // (H + 2*HKV) * D
#define HD_  4096   // H * D
#define SCALE_ 0.125f
#define NEG_ (-1.0e30f)
#define LOG2_THETA 17.1946011f   // log2(150000)

static __device__ __forceinline__ v8f wmma_bf16(const v16bf& a, const v16bf& b, v8f c) {
    return __builtin_amdgcn_wmma_f32_16x16x32_bf16(false, a, false, b, (short)0, c,
                                                   false, false);
}

// CDNA5 async DMA: global memory -> LDS, 16B per lane, tracked by ASYNCcnt.
static __device__ __forceinline__ void async_load_b128(unsigned lds_addr,
                                                       const void* gptr) {
    asm volatile("global_load_async_to_lds_b128 %0, %1, off"
                 :: "v"(lds_addr),
                    "v"((unsigned long long)(uintptr_t)gptr)
                 : "memory");
}
static __device__ __forceinline__ void wait_async_le8() {
    asm volatile("s_wait_asynccnt 8" ::: "memory");
}
static __device__ __forceinline__ void wait_async_0() {
    asm volatile("s_wait_asynccnt 0" ::: "memory");
}

// ---------------------------------------------------------------------------
// f32 -> bf16 (same layout)
// ---------------------------------------------------------------------------
__global__ __launch_bounds__(256) void cvt_bf16_kernel(const float* __restrict__ in,
                                                       __bf16* __restrict__ out, int n) {
    int i = (blockIdx.x * 256 + threadIdx.x) * 4;
    if (i < n) {
        float4 v = *(const float4*)(in + i);
        out[i + 0] = (__bf16)v.x;
        out[i + 1] = (__bf16)v.y;
        out[i + 2] = (__bf16)v.z;
        out[i + 3] = (__bf16)v.w;
    }
}

// ---------------------------------------------------------------------------
// f32 [K][N] -> bf16 [N][K]  (tiled transpose through LDS)
// ---------------------------------------------------------------------------
__global__ __launch_bounds__(256) void cvt_transpose_kernel(const float* __restrict__ in,
                                                            __bf16* __restrict__ out,
                                                            int K, int N) {
    __shared__ float tile[32][33];
    int kb = blockIdx.x * 32, nb = blockIdx.y * 32;
    int tx = threadIdx.x & 31, ty = threadIdx.x >> 5;  // ty: 0..7
#pragma unroll
    for (int i = ty; i < 32; i += 8) {
        int k = kb + i, n = nb + tx;
        tile[i][tx] = (k < K && n < N) ? in[(size_t)k * N + n] : 0.0f;
    }
    __syncthreads();
#pragma unroll
    for (int i = ty; i < 32; i += 8) {
        int n = nb + i, k = kb + tx;
        if (n < N && k < K) out[(size_t)n * K + k] = (__bf16)tile[tx][i];
    }
}

// ---------------------------------------------------------------------------
// Generic bf16 GEMM:  C[2048][N] (f32) = A[2048][K] (bf16) @ BT[N][K] (bf16) + bias
// Block tile 128x128x64, 8 waves, each wave 64x32 = 4x2 WMMA frags.
// Double-buffered LDS staged with async global->LDS DMA.
// ---------------------------------------------------------------------------
__global__ __launch_bounds__(256) void gemm_bf16_kernel(const __bf16* __restrict__ A,
                                                        const __bf16* __restrict__ BT,
                                                        const float* __restrict__ bias,
                                                        float* __restrict__ C,
                                                        int K, int N) {
    constexpr int LD = 72;  // padded bf16 stride (avoids LDS bank conflicts)
    __shared__ __bf16 As[2][128 * LD];
    __shared__ __bf16 Bs[2][128 * LD];

    const int tid  = threadIdx.x;
    const int lane = tid & 31;
    const int wave = tid >> 5;
    const int wr = wave >> 2;            // 0..1 -> 64-row slab
    const int wc = wave & 3;             // 0..3 -> 32-col slab
    const int m0 = blockIdx.y * 128;
    const int n0 = blockIdx.x * 128;
    const int row16 = lane & 15;
    const int kb    = (lane >> 4) * 8;   // WMMA 16-bit A/B layout half-select

    // Issue one 128x64 bf16 tile pair fill (8 async b128 loads per thread).
    auto issue_fill = [&](int buf, int k0) {
#pragma unroll
        for (int it = 0; it < 4; ++it) {
            int c = it * 256 + tid;
            int r = c >> 3, j = (c & 7) * 8;
            async_load_b128((unsigned)(uintptr_t)&As[buf][r * LD + j],
                            &A[(size_t)(m0 + r) * K + k0 + j]);
            int nrow = n0 + r;
            if (nrow >= N) nrow = N - 1;  // clamp; epilogue guards the store
            async_load_b128((unsigned)(uintptr_t)&Bs[buf][r * LD + j],
                            &BT[(size_t)nrow * K + k0 + j]);
        }
    };

    v8f acc[4][2];
#pragma unroll
    for (int mi = 0; mi < 4; ++mi)
#pragma unroll
        for (int ni = 0; ni < 2; ++ni)
            acc[mi][ni] = (v8f){0.f, 0.f, 0.f, 0.f, 0.f, 0.f, 0.f, 0.f};

    issue_fill(0, 0);
    for (int k0 = 0; k0 < K; k0 += 64) {
        const int buf = (k0 >> 6) & 1;
        if (k0 + 64 < K) {
            issue_fill(buf ^ 1, k0 + 64);  // prefetch next stage
            wait_async_le8();              // in-order: current stage landed
        } else {
            wait_async_0();
        }
        __syncthreads();

#pragma unroll
        for (int kk = 0; kk < 64; kk += 32) {
            Frag bf[2];
#pragma unroll
            for (int ni = 0; ni < 2; ++ni) {
                const __bf16* p = &Bs[buf][(wc * 32 + ni * 16 + row16) * LD + kk + kb];
                bf[ni].u[0] = *(const uint4*)p;
                bf[ni].u[1] = *(const uint4*)(p + 16);
            }
#pragma unroll
            for (int mi = 0; mi < 4; ++mi) {
                Frag af;
                const __bf16* p = &As[buf][(wr * 64 + mi * 16 + row16) * LD + kk + kb];
                af.u[0] = *(const uint4*)p;
                af.u[1] = *(const uint4*)(p + 16);
#pragma unroll
                for (int ni = 0; ni < 2; ++ni)
                    acc[mi][ni] = wmma_bf16(af.v, bf[ni].v, acc[mi][ni]);
            }
        }
        __syncthreads();  // all waves done with buf before it is refilled
    }

    // epilogue: C-layout lane holds col = row16, rows r + 8*(lane>>4)
#pragma unroll
    for (int mi = 0; mi < 4; ++mi)
#pragma unroll
        for (int ni = 0; ni < 2; ++ni) {
            int col = n0 + wc * 32 + ni * 16 + row16;
            if (col < N) {
                float b = bias[col];
#pragma unroll
                for (int r = 0; r < 8; ++r) {
                    int rowm = m0 + wr * 64 + mi * 16 + r + 8 * (lane >> 4);
                    C[(size_t)rowm * N + col] = acc[mi][ni][r] + b;
                }
            }
        }
}

// ---------------------------------------------------------------------------
// RoPE + split/convert:  qkv f32 [S][5120] ->
//   Qb bf16 [S][64][64] (roped), Kb bf16 [S][8][64] (roped),
//   Vt bf16 [8][64][S]  (transposed so V^T WMMA A-frags are contiguous)
// ---------------------------------------------------------------------------
__global__ __launch_bounds__(256) void rope_kernel(const float* __restrict__ qkv,
                                                   const int* __restrict__ positions,
                                                   __bf16* __restrict__ Qb,
                                                   __bf16* __restrict__ Kb,
                                                   __bf16* __restrict__ Vt) {
    const int s = blockIdx.x, t = threadIdx.x;
    const float pos = (float)positions[s];
    const float l2t = LOG2_THETA * (1.0f / 32.0f);
    const float* base = qkv + (size_t)s * NQKV;

    // Q: 64 heads * 32 pairs = 2048 pairs -> 8 per thread
#pragma unroll
    for (int j = 0; j < 8; ++j) {
        int p = j * 256 + t;
        int hh = p >> 5, i = p & 31;
        float ang = pos * exp2f(-(float)i * l2t);
        float c = __cosf(ang), sn = __sinf(ang);
        float x1 = base[hh * 64 + i];
        float x2 = base[hh * 64 + i + 32];
        size_t o = ((size_t)s * H_ + hh) * D_;
        Qb[o + i]      = (__bf16)(x1 * c - x2 * sn);
        Qb[o + i + 32] = (__bf16)(x2 * c + x1 * sn);
    }
    // K: 8 heads * 32 pairs = 256 pairs -> 1 per thread
    {
        int kh = t >> 5, i = t & 31;
        float ang = pos * exp2f(-(float)i * l2t);
        float c = __cosf(ang), sn = __sinf(ang);
        float x1 = base[HD_ + kh * 64 + i];
        float x2 = base[HD_ + kh * 64 + i + 32];
        size_t o = ((size_t)s * HKV_ + kh) * D_;
        Kb[o + i]      = (__bf16)(x1 * c - x2 * sn);
        Kb[o + i + 32] = (__bf16)(x2 * c + x1 * sn);
    }
    // V: 512 elems -> 2 per thread, transposed to [kh][d][s]
#pragma unroll
    for (int j = 0; j < 2; ++j) {
        int e = j * 256 + t;
        int kh = e >> 6, d = e & 63;
        Vt[((size_t)kh * D_ + d) * S_ + s] = (__bf16)base[HD_ + HKV_ * D_ + e];
    }
}

// ---------------------------------------------------------------------------
// Sliding-window attention with sinks.
// Grid: (S/16, HKV). Block: 256 (8 waves, one query head per wave).
// Each wave: 16 queries x one head; 9 key tiles of 16 covering [s0-128, s0+15].
// S^T = K @ Q^T via WMMA (softmax reduction lane-local); P^T maps lane-locally
// from C-layout to B-frag layout; O^T = V^T @ P^T via WMMA.
// K/V window staged into LDS with async global->LDS DMA.
// ---------------------------------------------------------------------------
__global__ __launch_bounds__(256) void attn_kernel(const __bf16* __restrict__ Qb,
                                                   const __bf16* __restrict__ Kb,
                                                   const __bf16* __restrict__ Vt,
                                                   const float* __restrict__ sinks,
                                                   __bf16* __restrict__ attnb) {
    constexpr int LDK = 72;    // Ks stride (bf16)
    constexpr int LDV = 160;   // Vts stride (bf16), 144 keys padded
    __shared__ __bf16 Ks[144 * LDK];   // [key in window][d]
    __shared__ __bf16 Vts[64 * LDV];   // [d][key in window]

    const int s0   = blockIdx.x * 16;
    const int kh   = blockIdx.y;
    const int tid  = threadIdx.x;
    const int lane = tid & 31;
    const int wave = tid >> 5;
    const int h    = kh * 8 + wave;     // query head for this wave
    const int qn   = lane & 15;         // query column (C/B layout)
    const int kb   = (lane >> 4) * 8;   // half-wave k-select
    const int qpos = s0 + qn;

    // ---- cooperative window fill (144 keys) via async DMA ----
#pragma unroll
    for (int it = 0; it < 5; ++it) {
        int c = it * 256 + tid;
        if (c < 1152) {
            // K tile: 144 rows x 8 chunks of 8 bf16
            int key = c >> 3, off8 = (c & 7) * 8;
            int gk = s0 - 128 + key;
            if (gk < 0) gk = 0;  // masked later; keeps reads in-bounds & finite
            async_load_b128((unsigned)(uintptr_t)&Ks[key * LDK + off8],
                            &Kb[((size_t)gk * HKV_ + kh) * D_ + off8]);
        }
    }
#pragma unroll
    for (int it = 0; it < 5; ++it) {
        int c = it * 256 + tid;
        if (c < 1152) {
            // V^T tile: 64 d-rows x 18 chunks of 8 keys
            int d = c / 18, j = c - d * 18;
            int koff = j * 8;
            int gk = s0 - 128 + koff;
            if (gk < 0) gk = 0;
            async_load_b128((unsigned)(uintptr_t)&Vts[d * LDV + koff],
                            &Vt[((size_t)kh * D_ + d) * S_ + gk]);
        }
    }
    wait_async_0();
    __syncthreads();

    // ---- Q fragments (B layout): lane holds query column qn, d runs kb..kb+7, kb+16..kb+23 ----
    Frag qf[2];
#pragma unroll
    for (int kk2 = 0; kk2 < 2; ++kk2) {
        const __bf16* p = &Qb[((size_t)(s0 + qn) * H_ + h) * D_ + kk2 * 32 + kb];
        qf[kk2].u[0] = *(const uint4*)p;
        qf[kk2].u[1] = *(const uint4*)(p + 16);
    }

    // ---- scores S^T for 9 key tiles ----
    float st[9][8];
#pragma unroll
    for (int t = 0; t < 9; ++t) {
        v8f c = (v8f){0.f, 0.f, 0.f, 0.f, 0.f, 0.f, 0.f, 0.f};
#pragma unroll
        for (int kk2 = 0; kk2 < 2; ++kk2) {
            Frag kf;
            const __bf16* p = &Ks[(t * 16 + (lane & 15)) * LDK + kk2 * 32 + kb];
            kf.u[0] = *(const uint4*)p;
            kf.u[1] = *(const uint4*)(p + 16);
            c = wmma_bf16(kf.v, qf[kk2].v, c);
        }
#pragma unroll
        for (int r = 0; r < 8; ++r) {
            int kg = s0 - 128 + t * 16 + r + 8 * (lane >> 4);  // key row (M)
            bool ok = (kg >= 0) && (kg <= qpos) && (kg >= qpos - (W_ - 1));
            st[t][r] = ok ? c[r] * SCALE_ : NEG_;
        }
    }

    // ---- softmax over 144 keys + sink (lane-local + one shfl) ----
    const float sk = sinks[h];
    float mx = sk;
#pragma unroll
    for (int t = 0; t < 9; ++t)
#pragma unroll
        for (int r = 0; r < 8; ++r) mx = fmaxf(mx, st[t][r]);
    mx = fmaxf(mx, __shfl_xor(mx, 16, 32));

    float sum = 0.0f;
#pragma unroll
    for (int t = 0; t < 9; ++t)
#pragma unroll
        for (int r = 0; r < 8; ++r) {
            float e = __expf(st[t][r] - mx);
            st[t][r] = e;
            sum += e;
        }
    sum += __shfl_xor(sum, 16, 32);
    sum += __expf(sk - mx);
    const float rcp = 1.0f / sum;

    // ---- O^T = V^T @ P^T : 5 key-tile pairs (last padded with zeros) ----
    v8f ot[4];
#pragma unroll
    for (int dc = 0; dc < 4; ++dc)
        ot[dc] = (v8f){0.f, 0.f, 0.f, 0.f, 0.f, 0.f, 0.f, 0.f};

#pragma unroll
    for (int tp = 0; tp < 5; ++tp) {
        Frag pb;  // P^T B-frag: 32 keys x 16 queries, lane-local repack
#pragma unroll
        for (int r = 0; r < 8; ++r) {
            pb.v[r]     = (__bf16)st[2 * tp][r];
            pb.v[8 + r] = (2 * tp + 1 < 9) ? (__bf16)st[2 * tp + 1][r] : (__bf16)0.0f;
        }
#pragma unroll
        for (int dc = 0; dc < 4; ++dc) {
            Frag vf;  // V^T A-frag: 16 d-rows x 32 keys
            const __bf16* p = &Vts[(dc * 16 + (lane & 15)) * LDV + tp * 32 + kb];
            vf.u[0] = *(const uint4*)p;
            vf.u[1] = *(const uint4*)(p + 16);
            ot[dc] = wmma_bf16(vf.v, pb.v, ot[dc]);
        }
    }

    // ---- normalize + store bf16 [s][h*64+d] for the output GEMM ----
#pragma unroll
    for (int dc = 0; dc < 4; ++dc)
#pragma unroll
        for (int r = 0; r < 8; ++r) {
            int d = dc * 16 + r + 8 * (lane >> 4);
            attnb[(size_t)(s0 + qn) * HD_ + h * D_ + d] = (__bf16)(ot[dc][r] * rcp);
        }
}

// ---------------------------------------------------------------------------
extern "C" void kernel_launch(void* const* d_in, const int* in_sizes, int n_in,
                              void* d_out, int out_size, void* d_ws, size_t ws_size,
                              hipStream_t stream) {
    const float* hidden    = (const float*)d_in[0];
    const int*   positions = (const int*)d_in[1];
    const float* w_qkv     = (const float*)d_in[2];
    const float* b_qkv     = (const float*)d_in[3];
    const float* w_o       = (const float*)d_in[4];
    const float* b_o       = (const float*)d_in[5];
    const float* sinks     = (const float*)d_in[6];
    float* out = (float*)d_out;

    size_t off = 0;
    auto carve = [&](size_t bytes) -> void* {
        void* p = (char*)d_ws + off;
        off += (bytes + 255) & ~(size_t)255;
        return p;
    };
    __bf16* hbf   = (__bf16*)carve((size_t)S_ * HID_ * 2);       // hidden bf16
    __bf16* wqkvT = (__bf16*)carve((size_t)NQKV * HID_ * 2);     // [5120][2880]
    __bf16* woT   = (__bf16*)carve((size_t)HID_ * HD_ * 2);      // [2880][4096]
    float*  qkv   = (float*) carve((size_t)S_ * NQKV * 4);       // f32 qkv
    __bf16* Qb    = (__bf16*)carve((size_t)S_ * H_ * D_ * 2);
    __bf16* Kb    = (__bf16*)carve((size_t)S_ * HKV_ * D_ * 2);
    __bf16* Vt    = (__bf16*)carve((size_t)HKV_ * D_ * S_ * 2);
    __bf16* attnb = (__bf16*)carve((size_t)S_ * HD_ * 2);

    // 1) hidden f32 -> bf16
    {
        int n = S_ * HID_;
        cvt_bf16_kernel<<<n / (256 * 4), 256, 0, stream>>>(hidden, hbf, n);
    }
    // 2) w_qkv [2880][5120] -> bf16 [5120][2880]
    cvt_transpose_kernel<<<dim3((HID_ + 31) / 32, (NQKV + 31) / 32), 256, 0, stream>>>(
        w_qkv, wqkvT, HID_, NQKV);
    // 3) w_o [4096][2880] -> bf16 [2880][4096]
    cvt_transpose_kernel<<<dim3((HD_ + 31) / 32, (HID_ + 31) / 32), 256, 0, stream>>>(
        w_o, woT, HD_, HID_);
    // 4) QKV GEMM: qkv = hidden @ w_qkv + b_qkv
    gemm_bf16_kernel<<<dim3(NQKV / 128, S_ / 128), 256, 0, stream>>>(
        hbf, wqkvT, b_qkv, qkv, HID_, NQKV);
    // 5) RoPE + split
    rope_kernel<<<S_, 256, 0, stream>>>(qkv, positions, Qb, Kb, Vt);
    // 6) sliding-window attention
    attn_kernel<<<dim3(S_ / 16, HKV_), 256, 0, stream>>>(Qb, Kb, Vt, sinks, attnb);
    // 7) output GEMM: out = attn @ w_o + b_o
    gemm_bf16_kernel<<<dim3((HID_ + 127) / 128, S_ / 128), 256, 0, stream>>>(
        attnb, woT, b_o, out, HD_, HID_);
}
